// GraphConstructor_75617194213663
// MI455X (gfx1250) — matrix operand order repaired and checked by reading, not compiled, not run on hardware
//
#include <hip/hip_runtime.h>
#include <hip/hip_bf16.h>
#include <math.h>

// ---------------- problem constants (from reference) ----------------
constexpr int S_  = 512;
constexpr int D_  = 1024;
constexpr int H_  = 16;
constexpr int L_  = 2;
constexpr int DN_ = 512;
constexpr int DE_ = 256;
constexpr int K_  = 3;
constexpr int DH_ = D_ / H_;         // 64
constexpr int E_  = S_ * K_;         // 1536
constexpr int KE_ = 2 * D_ + 1;      // 2049 (edge weight row stride)
constexpr int KE2_ = 2 * D_;         // 2048 (aligned GEMM K for edge MLP)

typedef __attribute__((ext_vector_type(16))) __bf16 v16bf;
typedef __attribute__((ext_vector_type(8)))  float  v8f;

__device__ __forceinline__ unsigned lds_off(const void* p) {
    // generic pointer into LDS aperture: addr[31:0] is the LDS byte offset
    return (unsigned)(unsigned long long)p;
}

// =====================================================================
// Batched GEMM (tile-aligned: M%128==0, N%64==0, K%32==0):
//   C[b,m,n] = act(scale * sum_k A[b,m,k]*W[b,n,k] + bias[n])
// f32 tiles DMA'd to double-buffered LDS via GLOBAL_LOAD_ASYNC_TO_LDS_B128
// (ASYNCcnt + split barriers), bf16 conversion fused into fragment build,
// V_WMMA_F32_16X16X32_BF16 accumulation.
// Block: 256 threads = 8 waves; C tile 128x64; wave -> 32x32 (4 WMMAs,
// 2 A-frags x 2 B-frags reused).
// =====================================================================
__global__ __launch_bounds__(256)
void k_gemm_fast_wmma(const float* __restrict__ A, long long bsA, int ldA,
                      const float* __restrict__ W, long long bsW, int ldW,
                      const float* __restrict__ bias,
                      float* __restrict__ C, long long bsC, int ldC,
                      int M, int N, int Kd, float scale, int act)
{
    __shared__ float Asf[2][128][32];   // 32 KB
    __shared__ float Bsf[2][64][32];    // 16 KB

    const int tid  = threadIdx.x;
    const int wave = tid >> 5;
    const int lane = tid & 31;
    const int wr   = wave & 3;      // 4 row groups  -> 4 x 32 rows
    const int wc   = wave >> 2;     // 2 col groups  -> 2 x 32 cols
    const int hf   = lane >> 4;
    const int l16  = lane & 15;

    const int tileM = blockIdx.y * 128;
    const int tileN = blockIdx.x * 64;

    const float* Ab = A + (size_t)blockIdx.z * bsA;
    const float* Wb = W + (size_t)blockIdx.z * bsW;
    float*       Cb = C + (size_t)blockIdx.z * bsC;

    // A tile 128x32 f32 = 1024 float4 chunks (4/thread);
    // B tile  64x32 f32 =  512 float4 chunks (2/thread).
    auto issue_stage = [&](int k0, int buf) {
        #pragma unroll
        for (int u = 0; u < 4; ++u) {
            int ch = tid + u * 256;
            int r = ch >> 3, c = (ch & 7) * 4;
            const float* g = Ab + (size_t)(tileM + r) * ldA + (k0 + c);
            unsigned l = lds_off(&Asf[buf][r][c]);
            asm volatile("global_load_async_to_lds_b128 %0, %1, off"
                         :: "v"(l), "v"(g) : "memory");
        }
        #pragma unroll
        for (int u = 0; u < 2; ++u) {
            int ch = tid + u * 256;
            int r = ch >> 3, c = (ch & 7) * 4;
            const float* g = Wb + (size_t)(tileN + r) * ldW + (k0 + c);
            unsigned l = lds_off(&Bsf[buf][r][c]);
            asm volatile("global_load_async_to_lds_b128 %0, %1, off"
                         :: "v"(l), "v"(g) : "memory");
        }
    };

    v8f acc[2][2] = {};
    const int nIter = Kd >> 5;           // Kd % 32 == 0

    issue_stage(0, 0);

    for (int it = 0; it < nIter; ++it) {
        const int cur = it & 1;
        // own wave's async transfers for buf[cur] complete:
        asm volatile("s_wait_asynccnt 0x0" ::: "memory");
        __syncthreads();                  // all waves' transfers visible; all
                                          // reads of buf[cur^1] finished
        if (it + 1 < nIter)
            issue_stage((it + 1) << 5, cur ^ 1);   // DMA overlaps compute below

        // ---- fragments (f32 LDS -> bf16), ISA VGPR layouts ----
        v16bf af[2], bf[2];
        const int ka = hf * 8;
        const int kb = hf * 16;
        #pragma unroll
        for (int mi = 0; mi < 2; ++mi) {
            const int ar = wr * 32 + mi * 16 + l16;
            #pragma unroll
            for (int j = 0; j < 8; ++j) {
                af[mi][j]     = (__bf16)Asf[cur][ar][ka + j];
                af[mi][8 + j] = (__bf16)Asf[cur][ar][16 + ka + j];
            }
        }
        #pragma unroll
        for (int ni = 0; ni < 2; ++ni) {
            const int br = wc * 32 + ni * 16 + l16;
            #pragma unroll
            for (int j = 0; j < 16; ++j)
                bf[ni][j] = (__bf16)Bsf[cur][br][(j < 8 ? ka + j : 8 + ka + j)];
        }
        #pragma unroll
        for (int mi = 0; mi < 2; ++mi)
            #pragma unroll
            for (int ni = 0; ni < 2; ++ni)
                acc[mi][ni] = __builtin_amdgcn_wmma_f32_16x16x32_bf16(
                    false, af[mi], false, bf[ni], (short)0, acc[mi][ni],
                    false, false);
    }

    // ---- epilogue: C frag layout: M = r + 8*hf, N = l16 ----
    #pragma unroll
    for (int mi = 0; mi < 2; ++mi) {
        #pragma unroll
        for (int ni = 0; ni < 2; ++ni) {
            #pragma unroll
            for (int r = 0; r < 8; ++r) {
                const int m = tileM + wr * 32 + mi * 16 + hf * 8 + r;
                const int n = tileN + wc * 32 + ni * 16 + l16;
                float v = acc[mi][ni][r] * scale + (bias ? bias[n] : 0.0f);
                if (act == 1)
                    v = 0.5f * v * (1.0f + erff(v * 0.70710678118654752f));
                Cb[(size_t)m * ldC + n] = v;
            }
        }
    }
    (void)M; (void)N;
}

// ============================ support kernels ============================

__global__ void k_embed(const int* __restrict__ tok, const float* __restrict__ emb,
                        const float* __restrict__ pe, float* __restrict__ x)
{
    int idx = blockIdx.x * blockDim.x + threadIdx.x;
    if (idx >= S_ * D_) return;
    int s = idx / D_, d = idx % D_;
    x[idx] = emb[(size_t)tok[s] * D_ + d] + pe[idx];
}

// qkv [S, 3D] -> q [H,S,DH], k [H,S,DH], vT [H,DH,S]
__global__ void k_split_qkv(const float* __restrict__ qkv, float* __restrict__ q,
                            float* __restrict__ k, float* __restrict__ vT)
{
    int idx = blockIdx.x * blockDim.x + threadIdx.x;
    if (idx >= H_ * S_ * DH_) return;
    int h = idx / (S_ * DH_);
    int r = idx % (S_ * DH_);
    int s = r / DH_, d = r % DH_;
    const float* row = qkv + (size_t)s * 3 * D_;
    q[idx] = row[h * DH_ + d];
    k[idx] = row[D_ + h * DH_ + d];
    vT[(size_t)h * DH_ * S_ + (size_t)d * S_ + s] = row[2 * D_ + h * DH_ + d];
}

// in-place row softmax over [rows, ncol]
__global__ __launch_bounds__(256)
void k_softmax_rows(float* __restrict__ p, int ncol)
{
    __shared__ float red[256];
    float* row = p + (size_t)blockIdx.x * ncol;
    int t = threadIdx.x;

    float mx = -1e30f;
    for (int j = t; j < ncol; j += 256) mx = fmaxf(mx, row[j]);
    red[t] = mx; __syncthreads();
    for (int s = 128; s > 0; s >>= 1) {
        if (t < s) red[t] = fmaxf(red[t], red[t + s]);
        __syncthreads();
    }
    mx = red[0]; __syncthreads();

    float sum = 0.0f;
    for (int j = t; j < ncol; j += 256) {
        float e = __expf(row[j] - mx);
        row[j] = e; sum += e;
    }
    red[t] = sum; __syncthreads();
    for (int s = 128; s > 0; s >>= 1) {
        if (t < s) red[t] += red[t + s];
        __syncthreads();
    }
    float inv = 1.0f / red[0];
    for (int j = t; j < ncol; j += 256) row[j] *= inv;
}

// attns_i[q,k] = mean over H of probs[h,q,k]
__global__ void k_head_avg(const float* __restrict__ probs, float* __restrict__ attn_i)
{
    int idx = blockIdx.x * blockDim.x + threadIdx.x;
    if (idx >= S_ * S_) return;
    float s = 0.0f;
    #pragma unroll
    for (int h = 0; h < H_; ++h) s += probs[(size_t)h * S_ * S_ + idx];
    attn_i[idx] = s * (1.0f / H_);
}

// out[s,:] = LayerNorm(x[s,:] + dlt[s,:]) * w + b
__global__ __launch_bounds__(256)
void k_add_layernorm(const float* __restrict__ x, const float* __restrict__ dlt,
                     const float* __restrict__ w, const float* __restrict__ b,
                     float* __restrict__ out)
{
    __shared__ float red[256];
    int s = blockIdx.x, t = threadIdx.x;
    const float* xr = x + (size_t)s * D_;
    const float* dr = dlt + (size_t)s * D_;
    float loc[D_ / 256];

    float sum = 0.0f;
    #pragma unroll
    for (int i = 0; i < D_ / 256; ++i) {
        loc[i] = xr[t + i * 256] + dr[t + i * 256];
        sum += loc[i];
    }
    red[t] = sum; __syncthreads();
    for (int st = 128; st > 0; st >>= 1) { if (t < st) red[t] += red[t + st]; __syncthreads(); }
    float mean = red[0] * (1.0f / D_); __syncthreads();

    float vs = 0.0f;
    #pragma unroll
    for (int i = 0; i < D_ / 256; ++i) { float dd = loc[i] - mean; vs += dd * dd; }
    red[t] = vs; __syncthreads();
    for (int st = 128; st > 0; st >>= 1) { if (t < st) red[t] += red[t + st]; __syncthreads(); }
    float inv = rsqrtf(red[0] * (1.0f / D_) + 1e-5f);

    #pragma unroll
    for (int i = 0; i < D_ / 256; ++i) {
        int j = t + i * 256;
        out[(size_t)s * D_ + j] = (loc[i] - mean) * inv * w[j] + b[j];
    }
}

// per-row top-3 of last-layer head-avg attention with zeroed diagonal
__global__ void k_topk_edges(const float* __restrict__ attn_last,
                             int* __restrict__ tgt, float* __restrict__ eidx)
{
    int q = blockIdx.x * blockDim.x + threadIdx.x;
    if (q >= S_) return;
    const float* row = attn_last + (size_t)q * S_;
    float v0 = -1e30f, v1 = -1e30f, v2 = -1e30f;
    int i0 = 0, i1 = 0, i2 = 0;
    for (int j = 0; j < S_; ++j) {
        float av = (j == q) ? 0.0f : row[j];
        if (av > v0)      { v2 = v1; i2 = i1; v1 = v0; i1 = i0; v0 = av; i0 = j; }
        else if (av > v1) { v2 = v1; i2 = i1; v1 = av; i1 = j; }
        else if (av > v2) { v2 = av; i2 = j; }
    }
    int idx3[3] = { i0, i1, i2 };
    #pragma unroll
    for (int j = 0; j < K_; ++j) {
        int e = q * K_ + j;
        tgt[e] = idx3[j];
        eidx[e]      = (float)q;
        eidx[E_ + e] = (float)idx3[j];
    }
}

// edge_in[e, c] (c < 2048): [lo[src], lo[tgt]]  -- attention column handled
// separately as a rank-1 fixup so the GEMM K stays tile-aligned.
__global__ void k_build_edge_in(const float* __restrict__ lo_i,
                                const int* __restrict__ tgt,
                                float* __restrict__ edge_in)
{
    int idx = blockIdx.x * blockDim.x + threadIdx.x;
    if (idx >= E_ * KE2_) return;
    int e = idx >> 11;          // / 2048
    int c = idx & (KE2_ - 1);   // % 2048
    int src = e / K_, tg = tgt[e];
    float v = (c < D_) ? lo_i[(size_t)src * D_ + c]
                       : lo_i[(size_t)tg * D_ + (c - D_)];
    edge_in[(size_t)e * KE2_ + c] = v;
}

// feats[e, n] += attn_i[src,tgt] * edge_w_i[n, 2048]   (the K=2049th column)
__global__ void k_edge_fixup(const float* __restrict__ attn_i,
                             const int* __restrict__ tgt,
                             const float* __restrict__ edgew_i,
                             float* __restrict__ feats_i)
{
    int idx = blockIdx.x * blockDim.x + threadIdx.x;
    if (idx >= E_ * DE_) return;
    int e = idx / DE_, n = idx % DE_;
    int src = e / K_;
    float a = attn_i[(size_t)src * S_ + tgt[e]];
    feats_i[(size_t)e * (DE_ * L_) + n] += a * edgew_i[(size_t)n * KE_ + KE2_];
}

// ============================== host launch ==============================
extern "C" void kernel_launch(void* const* d_in, const int* in_sizes, int n_in,
                              void* d_out, int out_size, void* d_ws, size_t ws_size,
                              hipStream_t stream)
{
    (void)in_sizes; (void)n_in; (void)out_size; (void)ws_size;

    const int*   tok   = (const int*)  d_in[0];
    const float* emb   = (const float*)d_in[1];
    const float* pe    = (const float*)d_in[2];
    const float* inw   = (const float*)d_in[3];
    const float* inb   = (const float*)d_in[4];
    const float* outw  = (const float*)d_in[5];
    const float* outb  = (const float*)d_in[6];
    const float* ln1w  = (const float*)d_in[7];
    const float* ln1b  = (const float*)d_in[8];
    const float* ln2w  = (const float*)d_in[9];
    const float* ln2b  = (const float*)d_in[10];
    const float* ff1w  = (const float*)d_in[11];
    const float* ff1b  = (const float*)d_in[12];
    const float* ff2w  = (const float*)d_in[13];
    const float* ff2b  = (const float*)d_in[14];
    const float* nodew = (const float*)d_in[15];
    const float* nodeb = (const float*)d_in[16];
    const float* edgew = (const float*)d_in[17];
    const float* edgeb = (const float*)d_in[18];
    const float* combw = (const float*)d_in[19];
    const float* combb = (const float*)d_in[20];

    char* wp = (char*)d_ws;
    auto alloc = [&](size_t nbytes) -> void* {
        void* p = (void*)wp;
        wp += (nbytes + 255) & ~(size_t)255;
        return p;
    };
    float* xf32   = (float*)alloc((size_t)S_ * D_ * 4);
    float* qkv    = (float*)alloc((size_t)S_ * 3 * D_ * 4);
    float* qh     = (float*)alloc((size_t)H_ * S_ * DH_ * 4);
    float* kh     = (float*)alloc((size_t)H_ * S_ * DH_ * 4);
    float* vT     = (float*)alloc((size_t)H_ * DH_ * S_ * 4);
    float* scores = (float*)alloc((size_t)H_ * S_ * S_ * 4);
    float* attns  = (float*)alloc((size_t)L_ * S_ * S_ * 4);
    float* ctx    = (float*)alloc((size_t)S_ * D_ * 4);
    float* proj   = (float*)alloc((size_t)S_ * D_ * 4);
    float* xtmp   = (float*)alloc((size_t)S_ * D_ * 4);
    float* ffo    = (float*)alloc((size_t)S_ * 4 * D_ * 4);
    float* ff2o   = (float*)alloc((size_t)S_ * D_ * 4);
    float* lo     = (float*)alloc((size_t)L_ * S_ * D_ * 4);
    float* edgein = (float*)alloc((size_t)E_ * KE2_ * 4);
    float* feats  = (float*)alloc((size_t)E_ * (DE_ * L_) * 4);
    int*   tgt    = (int*)  alloc((size_t)E_ * 4);

    float* out_node = (float*)d_out;
    float* out_edge = out_node + (size_t)S_ * DN_;
    float* out_eidx = out_edge + (size_t)E_ * DE_;

    auto cdiv = [](int a, int b) { return (a + b - 1) / b; };
    auto gemm = [&](const float* A, long long bsA, int ldA,
                    const float* W, long long bsW, int ldW,
                    const float* bias, float* C, long long bsC, int ldC,
                    int M, int N, int Kd, float scale, int act, int batch) {
        dim3 g(cdiv(N, 64), cdiv(M, 128), batch);
        k_gemm_fast_wmma<<<g, 256, 0, stream>>>(A, bsA, ldA, W, bsW, ldW, bias,
                                                C, bsC, ldC, M, N, Kd, scale, act);
    };

    k_embed<<<cdiv(S_ * D_, 256), 256, 0, stream>>>(tok, emb, pe, xf32);

    const float* xin = xf32;
    for (int i = 0; i < L_; ++i) {
        gemm(xin, 0, D_, inw + (size_t)i * 3 * D_ * D_, 0, D_,
             inb + (size_t)i * 3 * D_, qkv, 0, 3 * D_, S_, 3 * D_, D_, 1.0f, 0, 1);

        k_split_qkv<<<cdiv(H_ * S_ * DH_, 256), 256, 0, stream>>>(qkv, qh, kh, vT);

        gemm(qh, (long long)S_ * DH_, DH_, kh, (long long)S_ * DH_, DH_,
             nullptr, scores, (long long)S_ * S_, S_, S_, S_, DH_,
             1.0f / sqrtf((float)DH_), 0, H_);

        k_softmax_rows<<<H_ * S_, 256, 0, stream>>>(scores, S_);
        k_head_avg<<<cdiv(S_ * S_, 256), 256, 0, stream>>>(scores, attns + (size_t)i * S_ * S_);

        gemm(scores, (long long)S_ * S_, S_, vT, (long long)DH_ * S_, S_,
             nullptr, ctx, (long long)DH_, D_, S_, DH_, S_, 1.0f, 0, H_);

        gemm(ctx, 0, D_, outw + (size_t)i * D_ * D_, 0, D_,
             outb + (size_t)i * D_, proj, 0, D_, S_, D_, D_, 1.0f, 0, 1);

        k_add_layernorm<<<S_, 256, 0, stream>>>(xin, proj, ln1w + (size_t)i * D_,
                                                ln1b + (size_t)i * D_, xtmp);

        gemm(xtmp, 0, D_, ff1w + (size_t)i * 4 * D_ * D_, 0, D_,
             ff1b + (size_t)i * 4 * D_, ffo, 0, 4 * D_, S_, 4 * D_, D_, 1.0f, 1, 1);
        gemm(ffo, 0, 4 * D_, ff2w + (size_t)i * D_ * 4 * D_, 0, 4 * D_,
             ff2b + (size_t)i * D_, ff2o, 0, D_, S_, D_, 4 * D_, 1.0f, 0, 1);

        k_add_layernorm<<<S_, 256, 0, stream>>>(xtmp, ff2o, ln2w + (size_t)i * D_,
                                                ln2b + (size_t)i * D_, lo + (size_t)i * S_ * D_);
        xin = lo + (size_t)i * S_ * D_;
    }

    gemm(lo + (size_t)(L_ - 1) * S_ * D_, 0, D_, nodew, 0, D_,
         nodeb, out_node, 0, DN_, S_, DN_, D_, 1.0f, 0, 1);

    k_topk_edges<<<cdiv(S_, 256), 256, 0, stream>>>(attns + (size_t)(L_ - 1) * S_ * S_,
                                                    tgt, out_eidx);

    for (int i = 0; i < L_; ++i) {
        k_build_edge_in<<<cdiv(E_ * KE2_, 256), 256, 0, stream>>>(
            lo + (size_t)i * S_ * D_, tgt, edgein);
        // aligned K=2048 GEMM over [lo[src], lo[tgt]] part
        gemm(edgein, 0, KE2_, edgew + (size_t)i * DE_ * KE_, 0, KE_,
             edgeb + (size_t)i * DE_, feats + (size_t)i * DE_, 0, DE_ * L_,
             E_, DE_, KE2_, 1.0f, 0, 1);
        // rank-1 fixup for the attention column (K index 2048)
        k_edge_fixup<<<cdiv(E_ * DE_, 256), 256, 0, stream>>>(
            attns + (size_t)i * S_ * S_, tgt,
            edgew + (size_t)i * DE_ * KE_, feats + (size_t)i * DE_);
    }

    gemm(feats, 0, DE_ * L_, combw, 0, DE_ * L_,
         combb, out_edge, 0, DE_, E_, DE_, DE_ * L_, 1.0f, 0, 1);
}
